// Graph_Net_64939905516088
// MI455X (gfx1250) — matrix operand order repaired and checked by reading, not compiled, not run on hardware
//
#include <hip/hip_runtime.h>
#include <hip/hip_bf16.h>

#define Nn 20000
#define Ee 640000
#define Hh 512
#define Ll 4
#define Gg 256
#define Cc 10

typedef __attribute__((ext_vector_type(16))) __bf16 v16bf;
typedef __attribute__((ext_vector_type(8)))  float  v8f;

#if defined(__AMDGCN__) && __has_builtin(__builtin_amdgcn_global_load_async_to_lds_b128)
#define HAVE_ASYNC_LDS 1
// exact parameter types per hipcc diagnostic: v4i (vector_size 16 int) in AS1 / AS3
typedef int b128_t __attribute__((vector_size(16)));
typedef __attribute__((address_space(1))) b128_t* gb128p;
typedef __attribute__((address_space(3))) b128_t* lb128p;
#else
#define HAVE_ASYNC_LDS 0
#endif

#define WMMA_BF16(Aq, Bq, Cq) \
    __builtin_amdgcn_wmma_f32_16x16x32_bf16(false, (Aq), false, (Bq), (short)0, (Cq), false, false)

// ---------------- utility kernels ----------------

__global__ void fill_f32(float* __restrict__ p, float v, int n) {
    int i = blockIdx.x * blockDim.x + threadIdx.x;
    if (i < n) p[i] = v;
}

__global__ void deg_kernel(const int* __restrict__ ei, float* __restrict__ deg) {
    int e = blockIdx.x * blockDim.x + threadIdx.x;
    if (e < Ee) atomicAdd(&deg[ei[Ee + e]], 1.0f);
}

__global__ void rsqrt_kernel(float* __restrict__ deg) {
    int i = blockIdx.x * blockDim.x + threadIdx.x;
    if (i < Nn) deg[i] = rsqrtf(deg[i]);
}

__global__ void to_bf16(const float* __restrict__ in, __bf16* __restrict__ out, int n) {
    int i = blockIdx.x * blockDim.x + threadIdx.x;
    if (i < n) out[i] = (__bf16)in[i];
}

// WsT[l][j][k] = Ws[l][k][j] in bf16 so B-fragments are contiguous along K
__global__ void transposeW(const float* __restrict__ Ws, __bf16* __restrict__ WsT) {
    int idx = blockIdx.x * blockDim.x + threadIdx.x;
    if (idx >= Ll * Hh * Hh) return;
    int l   = idx >> 18;            // H*H = 2^18
    int rem = idx & ((1 << 18) - 1);
    int j   = rem >> 9;             // H = 2^9
    int k   = rem & (Hh - 1);
    WsT[idx] = (__bf16)Ws[(l << 18) + (k << 9) + j];
}

// ---------------- WMMA GEMM: C[N,512] = A[N,512](bf16) @ W (Bt = W^T, bf16) ----------------
// Block: 8 waves, 256 rows x 64 cols. B tile (64x512 bf16) staged in LDS via async-to-LDS,
// each wave computes 32 rows x 64 cols with double-buffered A fragments.
__global__ void __launch_bounds__(256) gemm_bf16(const __bf16* __restrict__ A,
                                                 const __bf16* __restrict__ Bt,
                                                 float* __restrict__ C) {
    constexpr int ROWSTR = 520;              // padded LDS row stride (bf16) -> conflict-free ds_load_b128
    __shared__ __bf16 Bs[64 * ROWSTR];       // ~65 KB of the 320 KB WGP LDS

    const int lane = threadIdx.x & 31;
    const int wave = threadIdx.x >> 5;
    const int j0   = blockIdx.y * 64;

    // ---- stage B tile: rows j0..j0+63 of Bt, all 512 K values ----
    for (int t = threadIdx.x; t < 64 * 64; t += 256) {   // 4096 x 16B
        int r = t >> 6;
        int c = t & 63;
        const float4* src = (const float4*)(Bt + (size_t)(j0 + r) * Hh) + c;
        float4*       dst = (float4*)(Bs + r * ROWSTR) + c;
#if HAVE_ASYNC_LDS
        __builtin_amdgcn_global_load_async_to_lds_b128((gb128p)src, (lb128p)dst, 0, 0);
#else
        *dst = *src;
#endif
    }
#if HAVE_ASYNC_LDS
#if __has_builtin(__builtin_amdgcn_s_wait_asynccnt)
    __builtin_amdgcn_s_wait_asynccnt(0);
#else
    asm volatile("s_wait_asynccnt 0x0" ::: "memory");
#endif
#endif
    __syncthreads();

    const int strip = blockIdx.x * 8 + wave;             // 32-row strip, 625 strips total
    if (strip >= Nn / 32) return;
    const int m0 = strip * 32;

    // A-frag (16x32): lanes 0-15 -> M=lane, K={0..7,16..23}; lanes 16-31 -> K={8..15,24..31}
    const int mr   = lane & 15;
    const int kh_a = (lane >> 4) << 3;   // 0 or 8
    // B-frag (32x16): lanes 0-15 -> N=lane, K={0..7,16..23}; lanes 16-31 -> K={8..15,24..31}
    const int nr   = lane & 15;
    const int kh_b = (lane >> 4) << 3;   // 0 or 8

    const __bf16* a0 = A + (size_t)(m0 + mr) * Hh + kh_a;        // rows m0..m0+15
    const __bf16* a1 = a0 + (size_t)16 * Hh;                     // rows m0+16..m0+31

    auto loadA = [&](v16bf& v, const __bf16* base, int k0) {
        ((float4*)&v)[0] = *(const float4*)(base + k0);
        ((float4*)&v)[1] = *(const float4*)(base + k0 + 16);
    };
    auto loadB = [&](v16bf& v, int t, int k0) {
        const __bf16* p = Bs + (t * 16 + nr) * ROWSTR + kh_b + k0;
        ((float4*)&v)[0] = *(const float4*)(p);
        ((float4*)&v)[1] = *(const float4*)(p + 16);
    };

    v8f acc[2][4] = {};
    v16bf a0f[2], a1f[2];
    loadA(a0f[0], a0, 0);
    loadA(a1f[0], a1, 0);

#pragma unroll
    for (int kk = 0; kk < 16; ++kk) {
        const int k0  = kk * 32;
        const int cur = kk & 1;
        const int nxt = cur ^ 1;
        if (kk < 15) {                       // prefetch next A fragments (overlaps the WMMAs)
            loadA(a0f[nxt], a0, k0 + 32);
            loadA(a1f[nxt], a1, k0 + 32);
        }
        v16bf b0, b1, b2, b3;
        loadB(b0, 0, k0);
        loadB(b1, 1, k0);
        loadB(b2, 2, k0);
        loadB(b3, 3, k0);
        acc[0][0] = WMMA_BF16(a0f[cur], b0, acc[0][0]);
        acc[0][1] = WMMA_BF16(a0f[cur], b1, acc[0][1]);
        acc[0][2] = WMMA_BF16(a0f[cur], b2, acc[0][2]);
        acc[0][3] = WMMA_BF16(a0f[cur], b3, acc[0][3]);
        acc[1][0] = WMMA_BF16(a1f[cur], b0, acc[1][0]);
        acc[1][1] = WMMA_BF16(a1f[cur], b1, acc[1][1]);
        acc[1][2] = WMMA_BF16(a1f[cur], b2, acc[1][2]);
        acc[1][3] = WMMA_BF16(a1f[cur], b3, acc[1][3]);
    }

    // C-frag: lane 0-15: VGPR r -> M=r; lanes 16-31: M=8+r; N = lane&15
    float* cp0 = C + (size_t)(m0 + ((lane >> 4) << 3)) * Hh + j0 + nr;
    float* cp1 = cp0 + (size_t)16 * Hh;
#pragma unroll
    for (int r = 0; r < 8; ++r) {
        cp0[(size_t)r * Hh +  0] = acc[0][0][r];
        cp0[(size_t)r * Hh + 16] = acc[0][1][r];
        cp0[(size_t)r * Hh + 32] = acc[0][2][r];
        cp0[(size_t)r * Hh + 48] = acc[0][3][r];
        cp1[(size_t)r * Hh +  0] = acc[1][0][r];
        cp1[(size_t)r * Hh + 16] = acc[1][1][r];
        cp1[(size_t)r * Hh + 32] = acc[1][2][r];
        cp1[(size_t)r * Hh + 48] = acc[1][3][r];
    }
}

// ---------------- normalized scatter-add aggregation (L2-resident atomics) ----------------
__global__ void __launch_bounds__(128) aggregate(const float* __restrict__ hw,
                                                 const int* __restrict__ ei,
                                                 const float* __restrict__ dis,
                                                 float* __restrict__ acc) {
    int e = blockIdx.x;
    int s, d;
    if (e < Ee) { s = ei[e]; d = ei[Ee + e]; }
    else        { s = d = e - Ee; }              // self-loops
    float norm = dis[s] * dis[d];
    const float4* row = (const float4*)(hw + (size_t)s * Hh);
    float* out = acc + (size_t)d * Hh;
    float4 v = row[threadIdx.x];
    int f = threadIdx.x * 4;
    atomicAdd(&out[f + 0], v.x * norm);
    atomicAdd(&out[f + 1], v.y * norm);
    atomicAdd(&out[f + 2], v.z * norm);
    atomicAdd(&out[f + 3], v.w * norm);
}

__global__ void bias_relu(const float* __restrict__ acc, const float* __restrict__ bias,
                          float* __restrict__ hid, __bf16* __restrict__ hbf) {
    int i = blockIdx.x * blockDim.x + threadIdx.x;   // N*H = 10,240,000 < 2^31
    int f = i & (Hh - 1);
    float v = fmaxf(acc[i] + bias[f], 0.0f);
    hid[i] = v;
    hbf[i] = (__bf16)v;
}

// ---------------- pooling ----------------
__global__ void count_kernel(const int* __restrict__ batch, float* __restrict__ counts) {
    int n = blockIdx.x * blockDim.x + threadIdx.x;
    if (n < Nn) atomicAdd(&counts[batch[n]], 1.0f);
}

__global__ void __launch_bounds__(256) pool_sum(const float* __restrict__ h,
                                                const int* __restrict__ batch,
                                                float* __restrict__ sums) {
    int n = blockIdx.x;
    int g = batch[n];
    const float* row = h + (size_t)n * Hh;
    float* out = sums + (size_t)g * Hh;
    for (int f = threadIdx.x; f < Hh; f += blockDim.x)
        atomicAdd(&out[f], row[f]);
}

__global__ void pooled_kernel(const float* __restrict__ sums, const float* __restrict__ counts,
                              float* __restrict__ pooled) {
    int i = blockIdx.x * blockDim.x + threadIdx.x;
    if (i >= Gg * Hh) return;
    int g = i >> 9;
    pooled[i] = sums[i] / fmaxf(counts[g], 1.0f);
}

// ---------------- FFN + softmax ----------------
__global__ void __launch_bounds__(320) ffn_kernel(const float* __restrict__ pooled,
                                                  const float* __restrict__ Wffn,
                                                  float* __restrict__ ffn) {
    int g = blockIdx.x;
    int wave = threadIdx.x >> 5;   // 10 waves, one class each
    int lane = threadIdx.x & 31;
    const float* p = pooled + (size_t)g * Hh;
    float partial = 0.0f;
    for (int k = lane; k < Hh; k += 32)
        partial += p[k] * Wffn[k * Cc + wave];
    for (int off = 16; off > 0; off >>= 1)
        partial += __shfl_down(partial, off, 32);
    if (lane == 0) ffn[g * Cc + wave] = fmaxf(partial, 0.0f);
}

__global__ void softmax_kernel(const float* __restrict__ ffn,
                               float* __restrict__ ls, float* __restrict__ sm) {
    int g = blockIdx.x * blockDim.x + threadIdx.x;
    if (g >= Gg) return;
    const float* f = ffn + g * Cc;
    float m = f[0];
#pragma unroll
    for (int c = 1; c < Cc; ++c) m = fmaxf(m, f[c]);
    float se = 0.0f;
#pragma unroll
    for (int c = 0; c < Cc; ++c) se += __expf(f[c] - m);
    float lse = __logf(se);
#pragma unroll
    for (int c = 0; c < Cc; ++c) {
        float l = f[c] - m - lse;
        ls[g * Cc + c] = l;
        sm[g * Cc + c] = __expf(l);
    }
}

// ---------------- launch ----------------
extern "C" void kernel_launch(void* const* d_in, const int* in_sizes, int n_in,
                              void* d_out, int out_size, void* d_ws, size_t ws_size,
                              hipStream_t stream) {
    const float* x     = (const float*)d_in[0];
    const int*   ei    = (const int*)  d_in[1];
    const int*   batch = (const int*)  d_in[2];
    const float* Ws    = (const float*)d_in[3];
    const float* bs    = (const float*)d_in[4];
    const float* Wffn  = (const float*)d_in[5];

    float* out    = (float*)d_out;
    float* hids   = out;                                   // [L,N,H]
    float* pooled = out + (size_t)Ll * Nn * Hh;            // [G,H]
    float* ffn    = pooled + (size_t)Gg * Hh;              // [G,C]
    float* ls     = ffn + Gg * Cc;                         // [G,C]
    float* sm     = ls + Gg * Cc;                          // [G,C]

    char* p = (char*)d_ws;
    auto carve = [&](size_t bytes) { char* r = p; p += (bytes + 255) & ~(size_t)255; return r; };
    float*  deg    = (float*) carve((size_t)Nn * 4);               // becomes dis after rsqrt
    __bf16* WsT    = (__bf16*)carve((size_t)Ll * Hh * Hh * 2);
    __bf16* hbf    = (__bf16*)carve((size_t)Nn * Hh * 2);
    float*  hw     = (float*) carve((size_t)Nn * Hh * 4);
    float*  acc    = (float*) carve((size_t)Nn * Hh * 4);
    float*  sums   = (float*) carve((size_t)Gg * Hh * 4);
    float*  counts = (float*) carve((size_t)Gg * 4);
    (void)ws_size; (void)in_sizes; (void)n_in; (void)out_size;

    const int NH = Nn * Hh;

    // degree with self-loops -> dis = rsqrt(deg)
    fill_f32<<<(Nn + 255) / 256, 256, 0, stream>>>(deg, 1.0f, Nn);
    deg_kernel<<<(Ee + 255) / 256, 256, 0, stream>>>(ei, deg);
    rsqrt_kernel<<<(Nn + 255) / 256, 256, 0, stream>>>(deg);

    // precision prep
    to_bf16<<<(NH + 255) / 256, 256, 0, stream>>>(x, hbf, NH);
    transposeW<<<(Ll * Hh * Hh + 255) / 256, 256, 0, stream>>>(Ws, WsT);

    for (int l = 0; l < Ll; ++l) {
        gemm_bf16<<<dim3((Nn / 32 + 7) / 8, Hh / 64), 256, 0, stream>>>(
            hbf, WsT + (size_t)l * Hh * Hh, hw);
        fill_f32<<<(NH + 255) / 256, 256, 0, stream>>>(acc, 0.0f, NH);
        aggregate<<<Ee + Nn, 128, 0, stream>>>(hw, ei, deg, acc);
        bias_relu<<<NH / 256, 256, 0, stream>>>(
            acc, bs + (size_t)l * Hh, hids + (size_t)l * Nn * Hh, hbf);
    }

    // global mean pool
    fill_f32<<<(Gg * Hh + 255) / 256, 256, 0, stream>>>(sums, 0.0f, Gg * Hh);
    fill_f32<<<1, 256, 0, stream>>>(counts, 0.0f, Gg);
    count_kernel<<<(Nn + 255) / 256, 256, 0, stream>>>(batch, counts);
    pool_sum<<<Nn, 256, 0, stream>>>(hids + (size_t)(Ll - 1) * Nn * Hh, batch, sums);
    pooled_kernel<<<(Gg * Hh + 255) / 256, 256, 0, stream>>>(sums, counts, pooled);

    // FFN + softmax
    ffn_kernel<<<Gg, 320, 0, stream>>>(pooled, Wffn, ffn);
    softmax_kernel<<<1, 256, 0, stream>>>(ffn, ls, sm);
}
